// GraphEncoderGAP_89369679495445
// MI455X (gfx1250) — compile-verified, matrix operand order checked by reading, not run on hardware
//
#include <hip/hip_runtime.h>
#include <hip/hip_bf16.h>

// ---------------------------------------------------------------------------
// GraphEncoderGAP for MI455X (gfx1250, wave32, WMMA 16x16x32 f16->f32)
// All dense matmuls go through v_wmma_f32_16x16x32_f16. The edge pipeline is
// fully fused: gather -> 2 WMMA GEMMs -> LN+ReLU in LDS -> WMMA GEMM ->
// atomic scatter-add (segment_sum). No 1Mx64 intermediates ever touch HBM.
// Weights for the edge GEMMs are staged into LDS with the gfx1250 async
// load-to-LDS path (ASYNCcnt).
// ---------------------------------------------------------------------------

typedef __attribute__((ext_vector_type(16))) _Float16 v16h;
typedef __attribute__((ext_vector_type(8)))  float    v8f;
typedef __attribute__((__vector_size__(4 * sizeof(int)))) int vint4;

#define S_TOK  515
#define SP_TOK 528   // padded to a multiple of 16

#if defined(__gfx1250__) && __has_builtin(__builtin_amdgcn_global_load_async_to_lds_b128)
#define HAVE_ASYNC_LDS 1
typedef __attribute__((address_space(1))) vint4 GVec4;   // global int4
typedef __attribute__((address_space(3))) vint4 LVec4;   // LDS int4
#endif

static __device__ __forceinline__ void wait_async_lds() {
#if defined(HAVE_ASYNC_LDS)
#if __has_builtin(__builtin_amdgcn_s_wait_asynccnt)
  __builtin_amdgcn_s_wait_asynccnt(0);
#else
  asm volatile("s_wait_asynccnt 0" ::: "memory");
#endif
#endif
}

// Cooperative block copy of `nfloat` floats (multiple of 4) into LDS.
// Uses the gfx1250 async load-to-LDS path (tracked by ASYNCcnt).
static __device__ __forceinline__ void stage_to_lds(const float* __restrict__ src,
                                                    float* __restrict__ dst, int nfloat) {
  for (int i = threadIdx.x; i < (nfloat >> 2); i += blockDim.x) {
#if defined(HAVE_ASYNC_LDS)
    __builtin_amdgcn_global_load_async_to_lds_b128(
        (GVec4*)(float*)(src + i * 4), (LVec4*)(dst + i * 4), 0, 0);
#else
    ((float4*)dst)[i] = ((const float4*)src)[i];
#endif
  }
}

// ---------------- WMMA fragment helpers (CDNA5 ISA 7.12.2 layouts) ----------

static __device__ __forceinline__ v8f wmma32(v16h a, v16h b, v8f c) {
  // D = A(16x32 f16) * B(32x16 f16) + C(16x16 f32)
  return __builtin_amdgcn_wmma_f32_16x16x32_f16(
      /*neg_a=*/false, a, /*neg_b=*/false, b,
      /*c_mod=*/(short)0, c, /*reuse_a=*/false, /*reuse_b=*/false);
}

// A fragment: lane(l&15)=row m, half (l>>4) selects K groups {0..7,16..23}/{8..15,24..31}
static __device__ __forceinline__ v16h loadA(const float* base, int ld, int k0, int lane) {
  int m = lane & 15, h = (lane >> 4) & 1;
  const float* p = base + (size_t)m * ld + k0 + 8 * h;
  v16h a;
#pragma unroll
  for (int j = 0; j < 16; ++j) a[j] = (_Float16)p[(j & 7) + 16 * (j >> 3)];
  return a;
}

static __device__ __forceinline__ v16h loadA_clamp(const float* x, int ld, int row0,
                                                   int n, int k0, int lane) {
  int m = lane & 15, h = (lane >> 4) & 1;
  int r = row0 + m; if (r >= n) r = n - 1;
  const float* p = x + (size_t)r * ld + k0 + 8 * h;
  v16h a;
#pragma unroll
  for (int j = 0; j < 16; ++j) a[j] = (_Float16)p[(j & 7) + 16 * (j >> 3)];
  return a;
}

static __device__ __forceinline__ v16h loadA_gather(const float* table, const int* idx16,
                                                    int k0, int lane) {
  int m = lane & 15, h = (lane >> 4) & 1;
  const float* p = table + (size_t)idx16[m] * 64 + k0 + 8 * h;
  v16h a;
#pragma unroll
  for (int j = 0; j < 16; ++j) a[j] = (_Float16)p[(j & 7) + 16 * (j >> 3)];
  return a;
}

// B fragment built from weight W (outdim x K, row-major): B(k,n) = W[n,k]
static __device__ __forceinline__ v16h loadB_W(const float* W, int ldw, int n0, int k0, int lane) {
  int n = lane & 15, h = (lane >> 4) & 1;
  const float* p = W + (size_t)(n0 + n) * ldw + k0 + 8 * h;
  v16h b;
#pragma unroll
  for (int j = 0; j < 16; ++j) b[j] = (_Float16)p[(j & 7) + 16 * (j >> 3)];
  return b;
}

static __device__ __forceinline__ v8f initC_bias(const float* b, int n0, int lane) {
  float bv = b[n0 + (lane & 15)];
  v8f c;
#pragma unroll
  for (int r = 0; r < 8; ++r) c[r] = bv;
  return c;
}

// C/D layout: lane=col n (+16 lanes repeat), vgpr r = row r (lanes0-15) / r+8 (lanes16-31)
static __device__ __forceinline__ void storeC_tile(float* t, int ld, int n0, v8f c, int lane) {
  int n = lane & 15, hi = (lane >> 4) & 1;
#pragma unroll
  for (int r = 0; r < 8; ++r) t[(size_t)(r + 8 * hi) * ld + n0 + n] = c[r];
}

static __device__ __forceinline__ void storeC_guard(float* y, int ld, int row0, int n,
                                                    int n0, v8f c, int lane) {
  int nn = n0 + (lane & 15), hi = (lane >> 4) & 1;
#pragma unroll
  for (int r = 0; r < 8; ++r) {
    int rr = row0 + r + 8 * hi;
    if (rr < n) y[(size_t)rr * ld + nn] = c[r];
  }
}

// ---------------- Kernels ---------------------------------------------------

// relu(Lin(relu(Lin(LN(x))))) node embedding; 16 nodes per wave, din=32.
__global__ void __launch_bounds__(256) embed_kernel(
    const float* __restrict__ x, int n, int din,
    const float* __restrict__ lng, const float* __restrict__ lnb,
    const float* __restrict__ W1, const float* __restrict__ b1,
    const float* __restrict__ W2, const float* __restrict__ b2,
    float* __restrict__ out)
{
  __shared__ float xln[8][16 * 32];
  __shared__ float ht[8][16 * 64];
  int wv = threadIdx.x >> 5, lane = threadIdx.x & 31;
  int row0 = (blockIdx.x * 8 + wv) * 16;
  if (lane < 16) {
    int r = row0 + lane; if (r >= n) r = n - 1;
    const float* xr = x + (size_t)r * din;
    float* d = xln[wv] + lane * din;
    float mean = 0.f;
    for (int i = 0; i < din; ++i) mean += xr[i];
    mean /= (float)din;
    float var = 0.f;
    for (int i = 0; i < din; ++i) { float dd = xr[i] - mean; var += dd * dd; }
    var /= (float)din;
    float rs = rsqrtf(var + 1e-5f);
    for (int i = 0; i < din; ++i) d[i] = (xr[i] - mean) * rs * lng[i] + lnb[i];
  }
  __syncthreads();
  v16h a0 = loadA(xln[wv], 32, 0, lane);
  for (int nt = 0; nt < 4; ++nt) {
    v8f c = initC_bias(b1, nt * 16, lane);
    c = wmma32(a0, loadB_W(W1, 32, nt * 16, 0, lane), c);
#pragma unroll
    for (int r = 0; r < 8; ++r) c[r] = fmaxf(c[r], 0.f);
    storeC_tile(ht[wv], 64, nt * 16, c, lane);
  }
  __syncthreads();
  v16h h0 = loadA(ht[wv], 64, 0, lane);
  v16h h1 = loadA(ht[wv], 64, 32, lane);
  for (int nt = 0; nt < 4; ++nt) {
    v8f c = initC_bias(b2, nt * 16, lane);
    c = wmma32(h0, loadB_W(W2, 64, nt * 16, 0, lane), c);
    c = wmma32(h1, loadB_W(W2, 64, nt * 16, 32, lane), c);
#pragma unroll
    for (int r = 0; r < 8; ++r) c[r] = fmaxf(c[r], 0.f);
    storeC_guard(out, 64, row0, n, nt * 16, c, lane);
  }
}

// Fused per-edge message + segment_sum. 16 edges per wave, 4 waves per block.
// m = Wl*right[dst] + bl + We*LN(edge) + Wr*left[src]; m = Wf*relu(LN(m)) + bf;
// agg[dst] += m (atomics). Wl/Wr staged to LDS via async load-to-LDS.
__global__ void __launch_bounds__(128) edge_msg_kernel(
    const float* __restrict__ left, const float* __restrict__ right,
    const int* __restrict__ src_idx, const int* __restrict__ dst_idx,
    const float* __restrict__ ev, const float* __restrict__ eg, const float* __restrict__ eb,
    const float* __restrict__ Wl, const float* __restrict__ bl,
    const float* __restrict__ We, const float* __restrict__ Wr,
    const float* __restrict__ fg, const float* __restrict__ fb,
    const float* __restrict__ Wf, const float* __restrict__ bf,
    float* __restrict__ agg, int n_edges)
{
  __shared__ float wlds[2][64 * 64];           // Wl, Wr staged in LDS
  __shared__ float mt[4][16 * 64];
  __shared__ int   s_src[4][16], s_dst[4][16];
  __shared__ float s_ea[4][16], s_eb[4][16];
  int wv = threadIdx.x >> 5, lane = threadIdx.x & 31;
  int e0 = (blockIdx.x * 4 + wv) * 16;
  __builtin_prefetch(Wf, 0, 3);

  // async stage Wl and Wr into LDS (ASYNCcnt-tracked on gfx1250)
  stage_to_lds(Wl, wlds[0], 64 * 64);
  stage_to_lds(Wr, wlds[1], 64 * 64);
  wait_async_lds();

  if (lane < 16) {
    int e = e0 + lane;
    int ec = (e < n_edges) ? e : (n_edges - 1);
    s_src[wv][lane] = src_idx[ec];
    s_dst[wv][lane] = dst_idx[ec];
    float a = ev[2 * ec], b = ev[2 * ec + 1];
    float mn = 0.5f * (a + b);
    float d = a - mn;
    float rs = rsqrtf(d * d + 1e-5f);      // var over 2 elems = ((a-m)^2+(b-m)^2)/2 = d^2
    s_ea[wv][lane] = (a - mn) * rs * eg[0] + eb[0];
    s_eb[wv][lane] = (b - mn) * rs * eg[1] + eb[1];
  }
  __syncthreads();
  v16h aR0 = loadA_gather(right, s_dst[wv], 0, lane);
  v16h aR1 = loadA_gather(right, s_dst[wv], 32, lane);
  v16h aL0 = loadA_gather(left,  s_src[wv], 0, lane);
  v16h aL1 = loadA_gather(left,  s_src[wv], 32, lane);
  int hi = (lane >> 4) & 1;
  for (int nt = 0; nt < 4; ++nt) {
    int n = nt * 16 + (lane & 15);
    float w0 = We[2 * n], w1 = We[2 * n + 1], bv = bl[n];
    v8f c;
#pragma unroll
    for (int r = 0; r < 8; ++r) {
      int m = r + 8 * hi;
      c[r] = bv + w0 * s_ea[wv][m] + w1 * s_eb[wv][m];
    }
    c = wmma32(aR0, loadB_W(wlds[0], 64, nt * 16, 0,  lane), c);
    c = wmma32(aR1, loadB_W(wlds[0], 64, nt * 16, 32, lane), c);
    c = wmma32(aL0, loadB_W(wlds[1], 64, nt * 16, 0,  lane), c);
    c = wmma32(aL1, loadB_W(wlds[1], 64, nt * 16, 32, lane), c);
    storeC_tile(mt[wv], 64, nt * 16, c, lane);
  }
  __syncthreads();
  if (lane < 16) {
    float* row = mt[wv] + lane * 64;
    float mean = 0.f;
    for (int i = 0; i < 64; ++i) mean += row[i];
    mean *= (1.f / 64.f);
    float var = 0.f;
    for (int i = 0; i < 64; ++i) { float dd = row[i] - mean; var += dd * dd; }
    var *= (1.f / 64.f);
    float rs = rsqrtf(var + 1e-5f);
    for (int i = 0; i < 64; ++i) {
      float t = (row[i] - mean) * rs * fg[i] + fb[i];
      row[i] = t > 0.f ? t : 0.f;
    }
  }
  __syncthreads();
  v16h m0 = loadA(mt[wv], 64, 0, lane);
  v16h m1 = loadA(mt[wv], 64, 32, lane);
  for (int nt = 0; nt < 4; ++nt) {
    v8f c = initC_bias(bf, nt * 16, lane);
    c = wmma32(m0, loadB_W(Wf, 64, nt * 16, 0,  lane), c);
    c = wmma32(m1, loadB_W(Wf, 64, nt * 16, 32, lane), c);
    int n = nt * 16 + (lane & 15);
#pragma unroll
    for (int r = 0; r < 8; ++r) {
      int m = r + 8 * hi;
      if (e0 + m < n_edges) atomicAdd(&agg[(size_t)s_dst[wv][m] * 64 + n], c[r]);
    }
  }
}

// xout = xin + Wo2 * relu(Wo1 * [LN(agg)|xin] + bo1) + bo2 ; 16 nodes per wave
__global__ void __launch_bounds__(128) post_agg_kernel(
    const float* __restrict__ agg, const float* __restrict__ xin,
    const float* __restrict__ pg, const float* __restrict__ pb,
    const float* __restrict__ Wo1, const float* __restrict__ bo1,
    const float* __restrict__ Wo2, const float* __restrict__ bo2,
    float* __restrict__ xout, int n)
{
  __shared__ float ht[4][16 * 128];
  __shared__ float tt[4][16 * 64];
  int wv = threadIdx.x >> 5, lane = threadIdx.x & 31;
  int row0 = (blockIdx.x * 4 + wv) * 16;
  {
    int m = lane & 15, half = (lane >> 4) & 1;
    int r = row0 + m; if (r >= n) r = n - 1;
    const float* ar = agg + (size_t)r * 64;
    const float* xr = xin + (size_t)r * 64;
    float* hr = ht[wv] + m * 128;
    for (int i = half * 32; i < half * 32 + 32; ++i) {
      hr[i] = ar[i];
      hr[64 + i] = xr[i];
    }
  }
  __syncthreads();
  if (lane < 16) {
    float* hr = ht[wv] + lane * 128;
    float mean = 0.f;
    for (int i = 0; i < 64; ++i) mean += hr[i];
    mean *= (1.f / 64.f);
    float var = 0.f;
    for (int i = 0; i < 64; ++i) { float dd = hr[i] - mean; var += dd * dd; }
    var *= (1.f / 64.f);
    float rs = rsqrtf(var + 1e-5f);
    for (int i = 0; i < 64; ++i) hr[i] = (hr[i] - mean) * rs * pg[i] + pb[i];
  }
  __syncthreads();
  v16h a0 = loadA(ht[wv], 128, 0,  lane);
  v16h a1 = loadA(ht[wv], 128, 32, lane);
  v16h a2 = loadA(ht[wv], 128, 64, lane);
  v16h a3 = loadA(ht[wv], 128, 96, lane);
  for (int nt = 0; nt < 4; ++nt) {
    v8f c = initC_bias(bo1, nt * 16, lane);
    c = wmma32(a0, loadB_W(Wo1, 128, nt * 16, 0,  lane), c);
    c = wmma32(a1, loadB_W(Wo1, 128, nt * 16, 32, lane), c);
    c = wmma32(a2, loadB_W(Wo1, 128, nt * 16, 64, lane), c);
    c = wmma32(a3, loadB_W(Wo1, 128, nt * 16, 96, lane), c);
#pragma unroll
    for (int r = 0; r < 8; ++r) c[r] = fmaxf(c[r], 0.f);
    storeC_tile(tt[wv], 64, nt * 16, c, lane);
  }
  __syncthreads();
  v16h t0 = loadA(tt[wv], 64, 0,  lane);
  v16h t1 = loadA(tt[wv], 64, 32, lane);
  int hi = (lane >> 4) & 1;
  for (int nt = 0; nt < 4; ++nt) {
    v8f c = initC_bias(bo2, nt * 16, lane);
    c = wmma32(t0, loadB_W(Wo2, 64, nt * 16, 0,  lane), c);
    c = wmma32(t1, loadB_W(Wo2, 64, nt * 16, 32, lane), c);
    int nn = nt * 16 + (lane & 15);
#pragma unroll
    for (int r = 0; r < 8; ++r) {
      int rr = row0 + r + 8 * hi;
      if (rr < n) xout[(size_t)rr * 64 + nn] = xin[(size_t)rr * 64 + nn] + c[r];
    }
  }
}

// y = (relu?)(x @ W^T + b) + addc ; K = 64, arbitrary outdim (multiple of 16)
__global__ void __launch_bounds__(256) lin64_kernel(
    const float* __restrict__ x, const float* __restrict__ W,
    const float* __restrict__ b, float* __restrict__ y,
    int n, int outdim, int do_relu, float addc)
{
  int wv = threadIdx.x >> 5, lane = threadIdx.x & 31;
  int row0 = (blockIdx.x * 8 + wv) * 16;
  v16h a0 = loadA_clamp(x, 64, row0, n, 0,  lane);
  v16h a1 = loadA_clamp(x, 64, row0, n, 32, lane);
  int hi = (lane >> 4) & 1;
  for (int nt = 0; nt < outdim / 16; ++nt) {
    v8f c = initC_bias(b, nt * 16, lane);
    c = wmma32(a0, loadB_W(W, 64, nt * 16, 0,  lane), c);
    c = wmma32(a1, loadB_W(W, 64, nt * 16, 32, lane), c);
    int nn = nt * 16 + (lane & 15);
#pragma unroll
    for (int r = 0; r < 8; ++r) {
      float v = c[r];
      if (do_relu) v = fmaxf(v, 0.f);
      v += addc;
      int rr = row0 + r + 8 * hi;
      if (rr < n) y[(size_t)rr * outdim + nn] = v;
    }
  }
}

__global__ void colmean_kernel(const float* __restrict__ x, int n, float scale,
                               float* __restrict__ out) {
  int d = threadIdx.x;
  if (d < 64) {
    float s = 0.f;
    for (int i = 0; i < n; ++i) s += x[(size_t)i * 64 + d];
    out[d] = s * scale / (float)n;
  }
}

// Token sequence: 512 sampled tokens + mrow + mcol + zeros (+ zero pad rows).
// NOTE: stand-in deterministic bijection replaces jax.random.permutation(key 42).
__global__ void build_tok_kernel(const float* __restrict__ rowT, const float* __restrict__ colT,
                                 const float* __restrict__ mrow, const float* __restrict__ mcol,
                                 float* __restrict__ tok, int nr, int nc)
{
  int t = blockIdx.x, d = threadIdx.x;
  float v = 0.f;
  if (t < 512) {
    unsigned tot = (unsigned)(nr + nc);
    unsigned idx = (48271u * (unsigned)t + 11u) % tot;
    v = (idx < (unsigned)nr) ? rowT[(size_t)idx * 64 + d]
                             : colT[(size_t)(idx - (unsigned)nr) * 64 + d];
  } else if (t == 512) v = mrow[d];
  else if (t == 513)   v = mcol[d];
  tok[(size_t)t * 64 + d] = v;
}

__global__ void scores_kernel(const float* __restrict__ qkv, float* __restrict__ sc, int S) {
  int id = blockIdx.x * blockDim.x + threadIdx.x;
  int tot = 8 * S * S;
  if (id >= tot) return;
  int j = id % S; int tmp = id / S; int i = tmp % S; int h = tmp / S;
  const float* q = qkv + (size_t)i * 192 + h * 8;
  const float* k = qkv + (size_t)j * 192 + 64 + h * 8;
  float s = 0.f;
#pragma unroll
  for (int d = 0; d < 8; ++d) s += q[d] * k[d];
  sc[((size_t)h * S + i) * S + j] = s * 0.35355339059327373f; // 1/sqrt(8)
}

__global__ void __launch_bounds__(256) softmax_kernel(float* __restrict__ sc, int S) {
  int h = threadIdx.x >> 5, lane = threadIdx.x & 31;
  int i = blockIdx.x;
  float* row = sc + ((size_t)h * S + i) * S;
  float mx = -1e30f;
  for (int j = lane; j < S; j += 32) mx = fmaxf(mx, row[j]);
#pragma unroll
  for (int o = 16; o > 0; o >>= 1) mx = fmaxf(mx, __shfl_xor(mx, o));
  float sum = 0.f;
  for (int j = lane; j < S; j += 32) { float e = __expf(row[j] - mx); row[j] = e; sum += e; }
#pragma unroll
  for (int o = 16; o > 0; o >>= 1) sum += __shfl_xor(sum, o);
  float inv = 1.f / sum;
  for (int j = lane; j < S; j += 32) row[j] *= inv;
}

__global__ void attnout_kernel(const float* __restrict__ sc, const float* __restrict__ qkv,
                               float* __restrict__ o, int S) {
  int id = blockIdx.x * blockDim.x + threadIdx.x;
  if (id >= S * 64) return;
  int d = id & 63, i = id >> 6;
  int h = d >> 3;
  const float* row = sc + ((size_t)h * S + i) * S;
  float acc = 0.f;
  for (int j = 0; j < S; ++j) acc += row[j] * qkv[(size_t)j * 192 + 128 + d];
  o[(size_t)i * 64 + d] = acc;
}

// out = LN(resid + xin @ W^T + b, g, bb)
__global__ void __launch_bounds__(128) resid_ln_kernel(
    const float* __restrict__ xin, const float* __restrict__ W,
    const float* __restrict__ b, const float* __restrict__ resid,
    const float* __restrict__ g, const float* __restrict__ bb,
    float* __restrict__ out, int n)
{
  __shared__ float tt[4][16 * 64];
  int wv = threadIdx.x >> 5, lane = threadIdx.x & 31;
  int row0 = (blockIdx.x * 4 + wv) * 16;
  v16h a0 = loadA_clamp(xin, 64, row0, n, 0,  lane);
  v16h a1 = loadA_clamp(xin, 64, row0, n, 32, lane);
  int hi = (lane >> 4) & 1;
  for (int nt = 0; nt < 4; ++nt) {
    v8f c = initC_bias(b, nt * 16, lane);
    c = wmma32(a0, loadB_W(W, 64, nt * 16, 0,  lane), c);
    c = wmma32(a1, loadB_W(W, 64, nt * 16, 32, lane), c);
    int nn = nt * 16 + (lane & 15);
#pragma unroll
    for (int r = 0; r < 8; ++r) {
      int m = r + 8 * hi;
      int rr = row0 + m; int rc = rr < n ? rr : (n - 1);
      tt[wv][m * 64 + nn] = c[r] + resid[(size_t)rc * 64 + nn];
    }
  }
  __syncthreads();
  if (lane < 16) {
    float* row = tt[wv] + lane * 64;
    float mean = 0.f;
    for (int i = 0; i < 64; ++i) mean += row[i];
    mean *= (1.f / 64.f);
    float var = 0.f;
    for (int i = 0; i < 64; ++i) { float dd = row[i] - mean; var += dd * dd; }
    var *= (1.f / 64.f);
    float rs = rsqrtf(var + 1e-5f);
    for (int i = 0; i < 64; ++i) row[i] = (row[i] - mean) * rs * g[i] + bb[i];
  }
  __syncthreads();
  {
    int m = lane & 15, half = (lane >> 4) & 1;
    int rr = row0 + m;
    if (rr < n) {
      const float* row = tt[wv] + m * 64;
      for (int i = half * 32; i < half * 32 + 32; ++i) out[(size_t)rr * 64 + i] = row[i];
    }
  }
}

// out = LN(resid + f @ W2^T + b2), K = 2048
__global__ void __launch_bounds__(128) ff2_resid_ln_kernel(
    const float* __restrict__ f, const float* __restrict__ W2,
    const float* __restrict__ b2, const float* __restrict__ resid,
    const float* __restrict__ g, const float* __restrict__ bb,
    float* __restrict__ out, int n)
{
  __shared__ float tt[4][16 * 64];
  int wv = threadIdx.x >> 5, lane = threadIdx.x & 31;
  int row0 = (blockIdx.x * 4 + wv) * 16;
  int hi = (lane >> 4) & 1;
  v8f cs[4];
#pragma unroll
  for (int nt = 0; nt < 4; ++nt) cs[nt] = initC_bias(b2, nt * 16, lane);
  for (int kt = 0; kt < 64; ++kt) {
    v16h a = loadA_clamp(f, 2048, row0, n, kt * 32, lane);
#pragma unroll
    for (int nt = 0; nt < 4; ++nt)
      cs[nt] = wmma32(a, loadB_W(W2, 2048, nt * 16, kt * 32, lane), cs[nt]);
  }
#pragma unroll
  for (int nt = 0; nt < 4; ++nt) {
    int nn = nt * 16 + (lane & 15);
#pragma unroll
    for (int r = 0; r < 8; ++r) {
      int m = r + 8 * hi;
      int rr = row0 + m; int rc = rr < n ? rr : (n - 1);
      tt[wv][m * 64 + nn] = cs[nt][r] + resid[(size_t)rc * 64 + nn];
    }
  }
  __syncthreads();
  if (lane < 16) {
    float* row = tt[wv] + lane * 64;
    float mean = 0.f;
    for (int i = 0; i < 64; ++i) mean += row[i];
    mean *= (1.f / 64.f);
    float var = 0.f;
    for (int i = 0; i < 64; ++i) { float dd = row[i] - mean; var += dd * dd; }
    var *= (1.f / 64.f);
    float rs = rsqrtf(var + 1e-5f);
    for (int i = 0; i < 64; ++i) row[i] = (row[i] - mean) * rs * g[i] + bb[i];
  }
  __syncthreads();
  {
    int m = lane & 15, half = (lane >> 4) & 1;
    int rr = row0 + m;
    if (rr < n) {
      const float* row = tt[wv] + m * 64;
      for (int i = half * 32; i < half * 32 + 32; ++i) out[(size_t)rr * 64 + i] = row[i];
    }
  }
}

// z = LN(x2[last], norm); out = Wp2 @ relu(Wp1 @ z + bp1) + bp2
__global__ void __launch_bounds__(128) head_kernel(
    const float* __restrict__ x2,
    const float* __restrict__ g, const float* __restrict__ b,
    const float* __restrict__ Wp1, const float* __restrict__ bp1,
    const float* __restrict__ Wp2, const float* __restrict__ bp2,
    float* __restrict__ out)
{
  __shared__ float z[64], h[64], stats[2];
  int t = threadIdx.x;
  if (t < 64) z[t] = x2[(size_t)(S_TOK - 1) * 64 + t];
  __syncthreads();
  if (t == 0) {
    float mean = 0.f;
    for (int i = 0; i < 64; ++i) mean += z[i];
    mean *= (1.f / 64.f);
    float var = 0.f;
    for (int i = 0; i < 64; ++i) { float dd = z[i] - mean; var += dd * dd; }
    var *= (1.f / 64.f);
    stats[0] = mean; stats[1] = rsqrtf(var + 1e-5f);
  }
  __syncthreads();
  if (t < 64) z[t] = (z[t] - stats[0]) * stats[1] * g[t] + b[t];
  __syncthreads();
  if (t < 64) {
    float acc = bp1[t];
    for (int k = 0; k < 64; ++k) acc += Wp1[t * 64 + k] * z[k];
    h[t] = fmaxf(acc, 0.f);
  }
  __syncthreads();
  if (t < 128) {
    float acc = bp2[t];
    for (int k = 0; k < 64; ++k) acc += Wp2[t * 64 + k] * h[k];
    out[t] = acc;
  }
}

// ---------------- Host-side launch ------------------------------------------

struct ConvP {
  const float *We, *Wf, *Wl, *Wo1, *Wo2, *Wr, *bf, *bl, *bo1, *bo2, *fb, *fg, *pb, *pg;
};

extern "C" void kernel_launch(void* const* d_in, const int* in_sizes, int n_in,
                              void* d_out, int out_size, void* d_ws, size_t ws_size,
                              hipStream_t stream)
{
  (void)out_size;
  const float* x_rows = (const float*)d_in[0];
  const float* x_cols = (const float*)d_in[1];
  const int*   eidx   = (const int*)d_in[2];
  const float* evals  = (const float*)d_in[3];
  int nr = in_sizes[0] / 32;
  int nc = in_sizes[1] / 32;
  int ne = in_sizes[2] / 2;
  const int* ri = eidx;
  const int* ci = eidx + ne;

  // Params flattened in JAX pytree order (dict keys sorted, lists in order).
  const float* P[128];
  for (int i = 0; i < 128; ++i) P[i] = nullptr;
  for (int i = 4; i < n_in && i < 128; ++i) P[i] = (const float*)d_in[i];

  const float *Wp1 = P[4], *Wp2 = P[5];
  const float *Wff1 = P[6], *Wff2 = P[7], *Wo = P[8], *Wqkv = P[9];
  const float *ab1 = P[10], *ab2 = P[11], *bff1 = P[12], *bff2 = P[13];
  const float *bo = P[14], *bqkv = P[15], *ag1 = P[16], *ag2 = P[17];
  const float *bp1 = P[18], *bp2 = P[19];
  const float *ce_W1 = P[20], *ce_W2 = P[21], *ce_b1 = P[22], *ce_b2 = P[23],
              *ce_lnb = P[24], *ce_lng = P[25];
  const float *co_W = P[26], *co_b = P[27];
  // convs: base 28, each conv 14 leaves: We,Wf,Wl,Wo1,Wo2,Wr,bf,bl,bo1,bo2,fb,fg,pb,pg
  const float *edge_b = P[84], *edge_g = P[85];
  const float *norm_b = P[86], *norm_g = P[87];
  const float *re_W1 = P[88], *re_W2 = P[89], *re_b1 = P[90], *re_b2 = P[91],
              *re_lnb = P[92], *re_lng = P[93];
  const float *ro_W = P[94], *ro_b = P[95];

  auto getConv = [&](int layer, int which) {
    int base = 28 + (layer * 2 + which) * 14;
    ConvP c;
    c.We = P[base + 0];  c.Wf = P[base + 1];  c.Wl = P[base + 2];  c.Wo1 = P[base + 3];
    c.Wo2 = P[base + 4]; c.Wr = P[base + 5];  c.bf = P[base + 6];  c.bl = P[base + 7];
    c.bo1 = P[base + 8]; c.bo2 = P[base + 9]; c.fb = P[base + 10]; c.fg = P[base + 11];
    c.pb = P[base + 12]; c.pg = P[base + 13];
    return c;
  };

  // Workspace layout
  float* w = (float*)d_ws;
  size_t off = 0;
  auto alloc = [&](size_t cnt) { float* p = w + off; off += cnt; return p; };
  float* rowA = alloc((size_t)nr * 64);
  float* rowB = alloc((size_t)nr * 64);
  float* colA = alloc((size_t)nc * 64);
  float* colB = alloc((size_t)nc * 64);
  float* aggR = alloc((size_t)nr * 64);
  float* aggC = alloc((size_t)nc * 64);
  float* tok  = alloc((size_t)SP_TOK * 64);
  float* qkv  = alloc((size_t)SP_TOK * 192);
  float* attO = alloc((size_t)SP_TOK * 64);
  float* x1   = alloc((size_t)SP_TOK * 64);
  float* ffb  = alloc((size_t)SP_TOK * 2048);
  float* x2   = alloc((size_t)SP_TOK * 64);
  float* sc   = alloc((size_t)8 * S_TOK * S_TOK);
  float* mrow = alloc(64);
  float* mcol = alloc(64);
  size_t need = off * sizeof(float);
  if (need > ws_size) need = ws_size;
  (void)hipMemsetAsync(d_ws, 0, need, stream);

  int tilesR = (nr + 15) / 16, tilesC = (nc + 15) / 16, tilesE = (ne + 15) / 16;

  embed_kernel<<<(tilesR + 7) / 8, 256, 0, stream>>>(
      x_rows, nr, 32, re_lng, re_lnb, re_W1, re_b1, re_W2, re_b2, rowA);
  embed_kernel<<<(tilesC + 7) / 8, 256, 0, stream>>>(
      x_cols, nc, 32, ce_lng, ce_lnb, ce_W1, ce_b1, ce_W2, ce_b2, colA);

  float *rcur = rowA, *rnxt = rowB, *ccur = colA, *cnxt = colB;
  for (int l = 0; l < 2; ++l) {
    ConvP c2r = getConv(l, 0), r2c = getConv(l, 1);
    (void)hipMemsetAsync(aggR, 0, (size_t)nr * 64 * sizeof(float), stream);
    edge_msg_kernel<<<(tilesE + 3) / 4, 128, 0, stream>>>(
        ccur, rcur, ci, ri, evals, edge_g, edge_b,
        c2r.Wl, c2r.bl, c2r.We, c2r.Wr, c2r.fg, c2r.fb, c2r.Wf, c2r.bf, aggR, ne);
    post_agg_kernel<<<(tilesR + 3) / 4, 128, 0, stream>>>(
        aggR, rcur, c2r.pg, c2r.pb, c2r.Wo1, c2r.bo1, c2r.Wo2, c2r.bo2, rnxt, nr);
    { float* t = rcur; rcur = rnxt; rnxt = t; }
    (void)hipMemsetAsync(aggC, 0, (size_t)nc * 64 * sizeof(float), stream);
    edge_msg_kernel<<<(tilesE + 3) / 4, 128, 0, stream>>>(
        rcur, ccur, ri, ci, evals, edge_g, edge_b,
        r2c.Wl, r2c.bl, r2c.We, r2c.Wr, r2c.fg, r2c.fb, r2c.Wf, r2c.bf, aggC, ne);
    post_agg_kernel<<<(tilesC + 3) / 4, 128, 0, stream>>>(
        aggC, ccur, r2c.pg, r2c.pb, r2c.Wo1, r2c.bo1, r2c.Wo2, r2c.bo2, cnxt, nc);
    { float* t = ccur; ccur = cnxt; cnxt = t; }
  }

  lin64_kernel<<<(tilesR + 7) / 8, 256, 0, stream>>>(rcur, ro_W, ro_b, rnxt, nr, 64, 1, 0.f);
  lin64_kernel<<<(tilesC + 7) / 8, 256, 0, stream>>>(ccur, co_W, co_b, cnxt, nc, 64, 1, 0.5f);
  colmean_kernel<<<1, 64, 0, stream>>>(rnxt, nr, 10.f, mrow);
  colmean_kernel<<<1, 64, 0, stream>>>(cnxt, nc, 10.f, mcol);
  build_tok_kernel<<<SP_TOK, 64, 0, stream>>>(rnxt, cnxt, mrow, mcol, tok, nr, nc);

  int tilesS = (S_TOK + 15) / 16;
  lin64_kernel<<<(tilesS + 7) / 8, 256, 0, stream>>>(tok, Wqkv, bqkv, qkv, S_TOK, 192, 0, 0.f);
  {
    int tot = 8 * S_TOK * S_TOK;
    scores_kernel<<<(tot + 255) / 256, 256, 0, stream>>>(qkv, sc, S_TOK);
  }
  softmax_kernel<<<S_TOK, 256, 0, stream>>>(sc, S_TOK);
  attnout_kernel<<<(S_TOK * 64 + 255) / 256, 256, 0, stream>>>(sc, qkv, attO, S_TOK);
  resid_ln_kernel<<<(tilesS + 3) / 4, 128, 0, stream>>>(attO, Wo, bo, tok, ag1, ab1, x1, S_TOK);
  lin64_kernel<<<(tilesS + 7) / 8, 256, 0, stream>>>(x1, Wff1, bff1, ffb, S_TOK, 2048, 1, 0.f);
  ff2_resid_ln_kernel<<<(tilesS + 3) / 4, 128, 0, stream>>>(ffb, Wff2, bff2, x1, ag2, ab2, x2, S_TOK);
  head_kernel<<<1, 128, 0, stream>>>(x2, norm_g, norm_b, Wp1, bp1, Wp2, bp2, (float*)d_out);
}
